// ResidualRecyclingEditor_23553600651863
// MI455X (gfx1250) — compile-verified, matrix operand order checked by reading, not run on hardware
//
#include <hip/hip_runtime.h>
#include <hip/hip_bf16.h>

typedef _Float16 v16h __attribute__((ext_vector_type(16)));
typedef float    v8f  __attribute__((ext_vector_type(8)));

union H16 { v16h v; _Float16 e[16]; };

constexpr int Dd     = 512;   // feature dim
constexpr int Rr     = 8;     // harmonic rank
constexpr int KSTEPS = 17;    // 16*32 = 512 feat K  +  1 step holding 8 coeff cols (+24 zero pad)
constexpr int MB     = 32;    // rows per block (2 WMMA m-tiles)
constexpr float EPSF = 1e-6f;

// ---------------- workspace layout (floats) ----------------
// [0..23]   segment sum(x,y,z)      [24..47] segment sum of squares
// [48..55]  segment counts
// [64..87]  segment mean            [88..111] segment 1/std
// [128..639] cvec  (bias + mean - mean@T^T)
// byte 4096: Bprep  (_Float16[17*32*32*16] = 544 KB, WMMA B-fragment layout)

__global__ void k_zero(float* __restrict__ ws) {
    if (threadIdx.x < 56) ws[threadIdx.x] = 0.f;
}

// ---- per-batch coordinate statistics: sum / sumsq / count --------------------
__global__ __launch_bounds__(256) void k_segred(const float* __restrict__ coord,
                                                const int* __restrict__ offset,
                                                float* __restrict__ ws, int Npts) {
    __shared__ float sS[24], sQ[24], sC[8];
    int tid = threadIdx.x;
    if (tid < 24) { sS[tid] = 0.f; sQ[tid] = 0.f; }
    if (tid < 8)  sC[tid] = 0.f;
    int off[8];
#pragma unroll
    for (int i = 0; i < 8; i++) off[i] = offset[i];
    __syncthreads();

    const int PER = 8;
    long base = ((long)blockIdx.x * 256 + tid) * PER;   // 8 consecutive points/thread
    int curb = -1;
    float ax = 0, ay = 0, az = 0, qx = 0, qy = 0, qz = 0, cnt = 0;
    for (int p = 0; p < PER; p++) {
        long n = base + p;
        if (n >= Npts) break;
        int b = 0;
#pragma unroll
        for (int i = 0; i < 8; i++) b += (off[i] <= n) ? 1 : 0;  // searchsorted 'right'
        if (b > 7) b = 7;
        if (b != curb) {
            if (curb >= 0) {
                atomicAdd(&sS[curb * 3 + 0], ax); atomicAdd(&sS[curb * 3 + 1], ay); atomicAdd(&sS[curb * 3 + 2], az);
                atomicAdd(&sQ[curb * 3 + 0], qx); atomicAdd(&sQ[curb * 3 + 1], qy); atomicAdd(&sQ[curb * 3 + 2], qz);
                atomicAdd(&sC[curb], cnt);
            }
            curb = b; ax = ay = az = qx = qy = qz = cnt = 0.f;
        }
        float x = coord[n * 3 + 0], y = coord[n * 3 + 1], z = coord[n * 3 + 2];
        ax += x; ay += y; az += z; qx += x * x; qy += y * y; qz += z * z; cnt += 1.f;
    }
    if (curb >= 0) {
        atomicAdd(&sS[curb * 3 + 0], ax); atomicAdd(&sS[curb * 3 + 1], ay); atomicAdd(&sS[curb * 3 + 2], az);
        atomicAdd(&sQ[curb * 3 + 0], qx); atomicAdd(&sQ[curb * 3 + 1], qy); atomicAdd(&sQ[curb * 3 + 2], qz);
        atomicAdd(&sC[curb], cnt);
    }
    __syncthreads();
    if (tid < 24) { atomicAdd(&ws[tid], sS[tid]); atomicAdd(&ws[24 + tid], sQ[tid]); }
    if (tid < 8)  atomicAdd(&ws[48 + tid], sC[tid]);
}

// ---- finalize stats + cvec[j] = bias[j] + mean[j] - sum_k T[j][k]*mean[k] ----
__global__ __launch_bounds__(512) void k_finalize(const float* __restrict__ transform,
                                                  const float* __restrict__ meanD,
                                                  const float* __restrict__ biasD,
                                                  float* __restrict__ ws) {
    __shared__ float mS[Dd];
    int tid = threadIdx.x;
    mS[tid] = meanD[tid];
    if (tid < 24) {
        int b = tid / 3;
        float cnt = fmaxf(ws[48 + b], 1.f);
        float mn  = ws[tid] / cnt;
        float var = ws[24 + tid] / cnt - mn * mn;
        ws[64 + tid] = mn;
        ws[88 + tid] = 1.f / fmaxf(sqrtf(fmaxf(var, 0.f)), EPSF);
    }
    __syncthreads();
    float acc = 0.f;
    const float* trow = transform + (size_t)tid * Dd;
    for (int k = 0; k < Dd; k++) acc = fmaf(trow[k], mS[k], acc);
    ws[128 + tid] = biasD[tid] + meanD[tid] - acc;
}

// ---- build B' = [ (T - I)^T ; harm_basis^T ; 0 ] in f16, fragment layout ----
// fragment (t,g): 32 lanes x 16 halves. lane: n = lane&15 (col j within group),
// koff = (lane>>4)*8. half h: v=h>>1, K = (v>>2)*16 + koff + (v&3)*2 + (h&1).
__global__ __launch_bounds__(256) void k_prepb(const float* __restrict__ transform,
                                               const float* __restrict__ harm,
                                               _Float16* __restrict__ Bprep) {
    int task = blockIdx.x * 256 + threadIdx.x;      // ((t*32+g)*32 + lane)
    if (task >= KSTEPS * 32 * 32) return;
    int lane = task & 31;
    int g    = (task >> 5) & 31;
    int t    = task >> 10;
    int j    = g * 16 + (lane & 15);
    int koff = (lane >> 4) * 8;
    H16 hv;
#pragma unroll
    for (int h = 0; h < 16; h++) {
        int v = h >> 1;
        int k = t * 32 + ((v >> 2) << 4) + koff + ((v & 3) << 1) + (h & 1);
        float val;
        if (k < Dd)            val = transform[(size_t)j * Dd + k] - ((j == k) ? 1.f : 0.f);
        else if (k < Dd + Rr)  val = harm[(size_t)j * Rr + (k - Dd)];
        else                   val = 0.f;
        hv.e[h] = (_Float16)val;
    }
    *(v16h*)&Bprep[(size_t)task * 16] = hv.v;
}

// ---- fused GEMM: out = feat + f16WMMA(feat, T-I) + coeff@harm^T + cvec -------
__global__ __launch_bounds__(256) void k_gemm(const float* __restrict__ feat,
                                              const float* __restrict__ coord,
                                              const int* __restrict__ offset,
                                              const float* __restrict__ gw,
                                              const float* __restrict__ gmean,
                                              const float* __restrict__ gstd,
                                              const float* __restrict__ ws,
                                              const _Float16* __restrict__ Bprep,
                                              float* __restrict__ out) {
    __shared__ alignas(32) _Float16 Af[2 * KSTEPS * 32 * 16];  // 34 KB, fragment layout
    __shared__ float coeffS[MB * Rr];
    int tid = threadIdx.x;
    long n0 = (long)blockIdx.x * MB;

    // per-row geometry coefficients (rows n0..n0+31)
    if (tid < MB) {
        long n = n0 + tid;
        int b = 0;
#pragma unroll
        for (int i = 0; i < 8; i++) b += (offset[i] <= n) ? 1 : 0;
        if (b > 7) b = 7;
        float x = (coord[n * 3 + 0] - ws[64 + b * 3 + 0]) * ws[88 + b * 3 + 0];
        float y = (coord[n * 3 + 1] - ws[64 + b * 3 + 1]) * ws[88 + b * 3 + 1];
        float z = (coord[n * 3 + 2] - ws[64 + b * 3 + 2]) * ws[88 + b * 3 + 2];
        float ge[9] = { x, y, z, x * x, y * y, z * z, x * y, y * z, z * x };
        float gn[9];
#pragma unroll
        for (int q = 0; q < 9; q++) gn[q] = (ge[q] - gmean[q]) / fmaxf(gstd[q], EPSF);
#pragma unroll
        for (int r = 0; r < 8; r++) {
            float a = 0.f;
#pragma unroll
            for (int q = 0; q < 9; q++) a = fmaf(gn[q], gw[q * 8 + r], a);
            coeffS[tid * 8 + r] = a;
        }
    }
    __syncthreads();

    // stage A tile (fp32 -> f16) directly in WMMA A-fragment layout
    for (int task = tid; task < 2 * KSTEPS * 32; task += 256) {
        int lane  = task & 31;
        int mt    = (task >> 5) & 1;
        int t     = task >> 6;
        int mr    = lane & 15;
        int khalf = lane >> 4;
        int row   = mt * 16 + mr;
        H16 hv;
        if (t < 16) {
            const float*  src = feat + (n0 + row) * (long)Dd + t * 32 + khalf * 8;
            const float4* s4  = (const float4*)src;
            float4 f0 = s4[0], f1 = s4[1], f2 = s4[4], f3 = s4[5];   // K = koff+0..7, koff+16..23
            hv.e[0]  = (_Float16)f0.x; hv.e[1]  = (_Float16)f0.y; hv.e[2]  = (_Float16)f0.z; hv.e[3]  = (_Float16)f0.w;
            hv.e[4]  = (_Float16)f1.x; hv.e[5]  = (_Float16)f1.y; hv.e[6]  = (_Float16)f1.z; hv.e[7]  = (_Float16)f1.w;
            hv.e[8]  = (_Float16)f2.x; hv.e[9]  = (_Float16)f2.y; hv.e[10] = (_Float16)f2.z; hv.e[11] = (_Float16)f2.w;
            hv.e[12] = (_Float16)f3.x; hv.e[13] = (_Float16)f3.y; hv.e[14] = (_Float16)f3.z; hv.e[15] = (_Float16)f3.w;
        } else {   // K-extension step: 8 coeff columns then zeros
#pragma unroll
            for (int h = 0; h < 16; h++)
                hv.e[h] = (khalf == 0 && h < 8) ? (_Float16)coeffS[row * 8 + h] : (_Float16)0.f;
        }
        *(v16h*)&Af[((mt * KSTEPS + t) * 32 + lane) * 16] = hv.v;
    }
    __syncthreads();

    int wave = tid >> 5;
    int lane = tid & 31;
    v8f acc[2][4];
#pragma unroll
    for (int mt = 0; mt < 2; mt++)
#pragma unroll
        for (int gi = 0; gi < 4; gi++)
            acc[mt][gi] = (v8f){ 0.f, 0.f, 0.f, 0.f, 0.f, 0.f, 0.f, 0.f };

    const v16h* aptr = (const v16h*)Af;
    for (int t = 0; t < KSTEPS; t++) {
        size_t bbase = (((size_t)t * 32 + wave * 4) * 32 + lane) * 16;
        v16h b0 = *(const v16h*)(Bprep + bbase);
        v16h b1 = *(const v16h*)(Bprep + bbase + 1 * 32 * 16);
        v16h b2 = *(const v16h*)(Bprep + bbase + 2 * 32 * 16);
        v16h b3 = *(const v16h*)(Bprep + bbase + 3 * 32 * 16);
        if (t + 1 < KSTEPS)
            __builtin_prefetch(Bprep + bbase + (size_t)32 * 32 * 16, 0, 0);  // next K-step, this wave's groups
        v16h a0 = aptr[(0 * KSTEPS + t) * 32 + lane];
        v16h a1 = aptr[(1 * KSTEPS + t) * 32 + lane];
        acc[0][0] = __builtin_amdgcn_wmma_f32_16x16x32_f16(false, a0, false, b0, (short)0, acc[0][0], false, false);
        acc[0][1] = __builtin_amdgcn_wmma_f32_16x16x32_f16(false, a0, false, b1, (short)0, acc[0][1], false, false);
        acc[0][2] = __builtin_amdgcn_wmma_f32_16x16x32_f16(false, a0, false, b2, (short)0, acc[0][2], false, false);
        acc[0][3] = __builtin_amdgcn_wmma_f32_16x16x32_f16(false, a0, false, b3, (short)0, acc[0][3], false, false);
        acc[1][0] = __builtin_amdgcn_wmma_f32_16x16x32_f16(false, a1, false, b0, (short)0, acc[1][0], false, false);
        acc[1][1] = __builtin_amdgcn_wmma_f32_16x16x32_f16(false, a1, false, b1, (short)0, acc[1][1], false, false);
        acc[1][2] = __builtin_amdgcn_wmma_f32_16x16x32_f16(false, a1, false, b2, (short)0, acc[1][2], false, false);
        acc[1][3] = __builtin_amdgcn_wmma_f32_16x16x32_f16(false, a1, false, b3, (short)0, acc[1][3], false, false);
    }

    // epilogue: D-tile layout — VGPR i: rows m = i + 8*(lane>>4), col = lane&15
    int jl = lane & 15;
    int mh = (lane >> 4) * 8;
#pragma unroll
    for (int mt = 0; mt < 2; mt++) {
#pragma unroll
        for (int gi = 0; gi < 4; gi++) {
            int j = (wave * 4 + gi) * 16 + jl;
            float cv = ws[128 + j];
            const float* fp = feat + (n0 + mt * 16 + mh) * (long)Dd + j;
            float*       op = out  + (n0 + mt * 16 + mh) * (long)Dd + j;
#pragma unroll
            for (int i = 0; i < 8; i++)
                op[(size_t)i * Dd] = acc[mt][gi][i] + fp[(size_t)i * Dd] + cv;
        }
    }
}

extern "C" void kernel_launch(void* const* d_in, const int* in_sizes, int n_in,
                              void* d_out, int out_size, void* d_ws, size_t ws_size,
                              hipStream_t stream) {
    const float* feat      = (const float*)d_in[0];
    const float* coord     = (const float*)d_in[1];
    const int*   offset    = (const int*)d_in[2];
    const float* transform = (const float*)d_in[3];
    const float* meanD     = (const float*)d_in[4];
    const float* biasD     = (const float*)d_in[5];
    const float* harm      = (const float*)d_in[6];
    const float* gw        = (const float*)d_in[7];
    const float* gmean     = (const float*)d_in[8];
    const float* gstd      = (const float*)d_in[9];
    float*       out       = (float*)d_out;
    float*       ws        = (float*)d_ws;
    _Float16*    Bprep     = (_Float16*)((char*)d_ws + 4096);

    int Npts = in_sizes[1] / 3;   // N = 524288

    k_zero<<<1, 64, 0, stream>>>(ws);
    int segBlocks = (Npts + 256 * 8 - 1) / (256 * 8);
    k_segred<<<segBlocks, 256, 0, stream>>>(coord, offset, ws, Npts);
    k_finalize<<<1, 512, 0, stream>>>(transform, meanD, biasD, ws);
    int prepTasks = KSTEPS * 32 * 32;
    k_prepb<<<(prepTasks + 255) / 256, 256, 0, stream>>>(transform, harm, Bprep);
    k_gemm<<<Npts / MB, 256, 0, stream>>>(feat, coord, offset, gw, gmean, gstd, ws, Bprep, out);
}
